// FrameTransformer_76733885710661
// MI455X (gfx1250) — matrix operand order repaired and checked by reading, not compiled, not run on hardware
//
#include <hip/hip_runtime.h>

// ---------------------------------------------------------------------------
// FrameTransformer block (per-channel multihead attention + FFN) for MI455X
// gfx1250.  All matmul work -> v_wmma_f32_16x16x32_f16 (wave32, f32 acc).
// Per-channel GEMMs: 64x128 block tiles, DOUBLE-BUFFERED async global->LDS
// staging (ASYNCcnt overlap), 32x32 per-wave tiles (4 WMMA/k-step).
// Attention scores: LDS-staged 64x128 S tiles (single k-step, dh=32).
// Softmax fused with P@V (b128 vectorized scans); the 512x512 probability
// tensor never hits HBM.
// ---------------------------------------------------------------------------

static constexpr int  Bb = 2;
static constexpr int  Cc = 16;
static constexpr int  Ff = 256;
static constexpr int  Wd = 512;
static constexpr int  Ee = 1024;
static constexpr int  Hh = 8;
static constexpr int  DH = Ff / Hh;                   // 32
static constexpr long S1 = (long)Bb * Cc * Ff * Wd;   // 4,194,304 elements

typedef __attribute__((ext_vector_type(16))) _Float16 v16h;
typedef __attribute__((ext_vector_type(8)))  float    v8f;

#define USE_ASYNC_LDS 1

__device__ __forceinline__ v8f wmma_f32_f16(v16h a, v16h b, v8f c) {
  return __builtin_amdgcn_wmma_f32_16x16x32_f16(false, a, false, b, (short)0, c,
                                                false, false);
}

// LDS byte offset of a __shared__ object: generic LDS addresses are
// aperture|offset, the low 32 bits are the LDS offset (ISA 10.2).
__device__ __forceinline__ unsigned lds_off(const void* p) {
  return (unsigned)(unsigned long long)p;
}

#if USE_ASYNC_LDS
// 16B global -> LDS async copy (GLOBAL_LOAD_ASYNC_TO_LDS_B128, ASYNCcnt).
__device__ __forceinline__ void async_cp_b128(unsigned lds, const void* g) {
  asm volatile("global_load_async_to_lds_b128 %0, %1, off"
               :: "v"(lds), "v"((unsigned long long)g) : "memory");
}
__device__ __forceinline__ void async_wait0() {
  asm volatile("s_wait_asynccnt 0x0" ::: "memory");
}
// Double-buffer wait: allow the 6 most recent copies (next tile) to remain
// outstanding; ASYNC ops complete in order, so the older buffer is done.
__device__ __forceinline__ void async_wait_le6() {
  asm volatile("s_wait_asynccnt 0x6" ::: "memory");
}
#endif

// ---------------------------------------------------------------------------
// Multichannel LayerNorm over the feature axis; one wave per (b,c,w) column.
// ---------------------------------------------------------------------------
__global__ void ln_kernel(const float* __restrict__ x,
                          const float* __restrict__ g,
                          const float* __restrict__ bta,
                          float* __restrict__ out) {
  const int wave = threadIdx.x >> 5;
  const int lane = threadIdx.x & 31;
  const long col = (long)blockIdx.x * 8 + wave;       // [0, B*C*W)
  const int  wpos = (int)(col % Wd);
  long t = col / Wd;
  const int cc2 = (int)(t % Cc);
  const int bb  = (int)(t / Cc);
  const float* base = x + (((long)bb * Cc + cc2) * Ff) * Wd + wpos;
  float sum = 0.f, sq = 0.f;
#pragma unroll
  for (int j = 0; j < Ff / 32; ++j) {
    const float v = base[(long)(lane + 32 * j) * Wd];
    sum += v; sq += v * v;
  }
#pragma unroll
  for (int off = 16; off >= 1; off >>= 1) {
    sum += __shfl_xor(sum, off, 32);
    sq  += __shfl_xor(sq,  off, 32);
  }
  const float mean = sum * (1.0f / Ff);
  const float var  = sq  * (1.0f / Ff) - mean * mean;
  const float inv  = rsqrtf(var + 1e-5f);
  float* ob = out + (((long)bb * Cc + cc2) * Ff) * Wd + wpos;
#pragma unroll
  for (int j = 0; j < Ff / 32; ++j) {
    const int f = lane + 32 * j;
    const float v = base[(long)f * Wd];
    ob[(long)f * Wd] = (v - mean) * inv * g[cc2 * Ff + f] + bta[cc2 * Ff + f];
  }
}

// ---------------------------------------------------------------------------
// 3x3 conv over the (f,w) plane, Cc->Cc channels, pad=1; weights in LDS.
// ---------------------------------------------------------------------------
__global__ void conv3x3_kernel(const float* __restrict__ in,
                               const float* __restrict__ wconv,
                               float* __restrict__ out) {
  __shared__ float wsm[Cc * Cc * 9];
  for (int i = threadIdx.x; i < Cc * Cc * 9; i += blockDim.x) wsm[i] = wconv[i];
  __syncthreads();
  const long idx = (long)blockIdx.x * blockDim.x + threadIdx.x;  // [0, S1)
  const int wpos = (int)(idx % Wd);
  long t = idx / Wd;
  const int fpos = (int)(t % Ff); t /= Ff;
  const int o  = (int)(t % Cc);
  const int bb = (int)(t / Cc);
  const float* bin = in + (long)bb * Cc * Ff * Wd;
  float acc = 0.0f;
  for (int ci = 0; ci < Cc; ++ci) {
    const float* pc  = bin + (long)ci * Ff * Wd;
    const float* wk9 = &wsm[(o * Cc + ci) * 9];
#pragma unroll
    for (int df = -1; df <= 1; ++df) {
      const int ff2 = fpos + df;
      if (ff2 < 0 || ff2 >= Ff) continue;
#pragma unroll
      for (int dw2 = -1; dw2 <= 1; ++dw2) {
        const int ww2 = wpos + dw2;
        if (ww2 < 0 || ww2 >= Wd) continue;
        acc += wk9[(df + 1) * 3 + (dw2 + 1)] * pc[(long)ff2 * Wd + ww2];
      }
    }
  }
  out[idx] = acc;
}

// ---------------------------------------------------------------------------
// Per-channel GEMM, double-buffered LDS tiling:
//   Y[bc] (MxN) = W[c] (MxK) * X[bc] (KxN)
// Block = 256 thr = 8 waves, 64x128 block tile; wave (2x4) owns 32x32
// (2x2 WMMA accumulators).  While computing k-step n from buffer n%2, the
// block async-copies k-step n+1 into buffer (n+1)%2 and partial-waits
// ASYNCcnt<=6 so the older buffer is guaranteed complete (in-order).
// ---------------------------------------------------------------------------
template <int M, int K, int N>
__global__ void gemm_pc_kernel(const float* __restrict__ Wt,
                               const float* __restrict__ X,
                               float* __restrict__ Y) {
  __shared__ float Wlds[2][64 * 32];    // [buf][row][k]   8KB each
  __shared__ float Xlds[2][32 * 128];   // [buf][k][col]  16KB each

  const int bc = blockIdx.y;            // b*Cc + c
  const int ch = bc % Cc;
  const float* Wc = Wt + (long)ch * M * K;
  const float* Xc = X  + (long)bc * K * N;
  float*       Yc = Y  + (long)bc * M * N;

  constexpr int NBn = N / 128;
  const int bM = (blockIdx.x / NBn) * 64;
  const int bN = (blockIdx.x % NBn) * 128;

  const int tid = threadIdx.x;
  const int wv  = tid >> 5;
  const int wm  = wv & 1;               // row offset wm*32
  const int wn  = wv >> 1;              // col offset wn*32
  const int lr  = tid & 15;
  const int hf  = (tid >> 4) & 1;

  auto stage = [&](int kk, int buf) {
    const unsigned wb = lds_off(&Wlds[buf][0]);
    const unsigned xb = lds_off(&Xlds[buf][0]);
#pragma unroll
    for (int i = 0; i < 2; ++i) {                  // W: 64x32 = 512 b128
      const int l = tid + i * 256;
      const int r = l >> 3, sg = (l & 7) << 2;
#if USE_ASYNC_LDS
      async_cp_b128(wb + (unsigned)(r * 32 + sg) * 4,
                    Wc + (long)(bM + r) * K + kk + sg);
#else
      *(float4*)&Wlds[buf][r * 32 + sg] =
          *(const float4*)(Wc + (long)(bM + r) * K + kk + sg);
#endif
    }
#pragma unroll
    for (int i = 0; i < 4; ++i) {                  // X: 32x128 = 1024 b128
      const int l = tid + i * 256;
      const int r = l >> 5, sg = (l & 31) << 2;
#if USE_ASYNC_LDS
      async_cp_b128(xb + (unsigned)(r * 128 + sg) * 4,
                    Xc + (long)(kk + r) * N + bN + sg);
#else
      *(float4*)&Xlds[buf][r * 128 + sg] =
          *(const float4*)(Xc + (long)(kk + r) * N + bN + sg);
#endif
    }
  };

  v8f acc00 = {}, acc01 = {}, acc10 = {}, acc11 = {};

  stage(0, 0);
  for (int kk = 0; kk < K; kk += 32) {
    const int  buf  = (kk >> 5) & 1;
    const bool more = (kk + 32 < K);
    if (more) stage(kk + 32, buf ^ 1);   // prefetch next tile (other buffer)
#if USE_ASYNC_LDS
    if (more) async_wait_le6();          // current buffer complete
    else      async_wait0();
#endif
    __syncthreads();

    v16h a0, a1, b0, b1;
    {
      const float* ba0 = &Wlds[buf][(wm * 32 + 0 + lr) * 32];
      const float* ba1 = &Wlds[buf][(wm * 32 + 16 + lr) * 32];
#pragma unroll
      for (int e = 0; e < 16; ++e) {
        const int k = hf * 16 + e;
        a0[e] = (_Float16)ba0[k];
        a1[e] = (_Float16)ba1[k];
      }
      const float* bb0 = &Xlds[buf][wn * 32 + 0 + lr];
      const float* bb1 = &Xlds[buf][wn * 32 + 16 + lr];
#pragma unroll
      for (int e = 0; e < 16; ++e) {
        const int k = hf * 16 + e;
        b0[e] = (_Float16)bb0[k * 128];
        b1[e] = (_Float16)bb1[k * 128];
      }
    }
    acc00 = wmma_f32_f16(a0, b0, acc00);
    acc01 = wmma_f32_f16(a0, b1, acc01);
    acc10 = wmma_f32_f16(a1, b0, acc10);
    acc11 = wmma_f32_f16(a1, b1, acc11);
    __syncthreads();                     // readers done before buf is reused
  }

  const int rbase = bM + wm * 32;
  const int cbase = bN + wn * 32;
#pragma unroll
  for (int r = 0; r < 8; ++r) {
    const int rr = r + 8 * hf;
    Yc[(long)(rbase + rr)      * N + cbase + lr]      = acc00[r];
    Yc[(long)(rbase + rr)      * N + cbase + 16 + lr] = acc01[r];
    Yc[(long)(rbase + 16 + rr) * N + cbase + lr]      = acc10[r];
    Yc[(long)(rbase + 16 + rr) * N + cbase + 16 + lr] = acc11[r];
  }
}

// ---------------------------------------------------------------------------
// Attention scores: S = (Q_h^T K_h)/sqrt(f) -> qk region of d_out.
// Block computes a 64(i) x 128(j) S tile; Q (32x64) and K (32x128) slabs are
// async-staged to LDS once (dh = 32 -> single k-step), then 8 waves do 2x2
// WMMAs each.  ~5x less L2 read traffic than per-tile global fragment loads.
// ---------------------------------------------------------------------------
__global__ void scores_kernel(const float* __restrict__ Q,
                              const float* __restrict__ Kmat,
                              float* __restrict__ S) {
  __shared__ float Qlds[32 * 64];       // [k][i]  8KB
  __shared__ float Klds[32 * 128];      // [k][j] 16KB

  const int bch = blockIdx.y;           // (b*Cc + c)*Hh + h
  const int hd  = bch & (Hh - 1);
  const int bc  = bch >> 3;
  const float* Qc = Q    + ((long)bc * Ff + (long)hd * DH) * Wd;
  const float* Kc = Kmat + ((long)bc * Ff + (long)hd * DH) * Wd;
  float* Sc = S + (long)bch * Wd * Wd;

  const int bI = (blockIdx.x >> 2) * 64;    // 8 i-blocks
  const int bJ = (blockIdx.x & 3) * 128;    // 4 j-blocks
  const int tid = threadIdx.x;

  {
    const unsigned qb = lds_off(Qlds);
    const unsigned kb = lds_off(Klds);
#pragma unroll
    for (int i = 0; i < 2; ++i) {           // Q: 32x64 = 512 b128
      const int l = tid + i * 256;
      const int r = l >> 4, sg = (l & 15) << 2;
#if USE_ASYNC_LDS
      async_cp_b128(qb + (unsigned)(r * 64 + sg) * 4, Qc + (long)r * Wd + bI + sg);
#else
      *(float4*)&Qlds[r * 64 + sg] = *(const float4*)(Qc + (long)r * Wd + bI + sg);
#endif
    }
#pragma unroll
    for (int i = 0; i < 4; ++i) {           // K: 32x128 = 1024 b128
      const int l = tid + i * 256;
      const int r = l >> 5, sg = (l & 31) << 2;
#if USE_ASYNC_LDS
      async_cp_b128(kb + (unsigned)(r * 128 + sg) * 4, Kc + (long)r * Wd + bJ + sg);
#else
      *(float4*)&Klds[r * 128 + sg] = *(const float4*)(Kc + (long)r * Wd + bJ + sg);
#endif
    }
#if USE_ASYNC_LDS
    async_wait0();
#endif
    __syncthreads();
  }

  const int wv = tid >> 5;
  const int wm = wv & 1;                    // i offset wm*32
  const int wn = wv >> 1;                   // j offset wn*32
  const int lr = tid & 15;
  const int hf = (tid >> 4) & 1;

  v16h a0, a1, b0, b1;
#pragma unroll
  for (int e = 0; e < 16; ++e) {
    const int k = hf * 16 + e;
    a0[e] = (_Float16)Qlds[k * 64 + wm * 32 + 0 + lr];
    a1[e] = (_Float16)Qlds[k * 64 + wm * 32 + 16 + lr];
  }
#pragma unroll
  for (int e = 0; e < 16; ++e) {
    const int k = hf * 16 + e;
    b0[e] = (_Float16)Klds[k * 128 + wn * 32 + 0 + lr];
    b1[e] = (_Float16)Klds[k * 128 + wn * 32 + 16 + lr];
  }
  v8f acc00 = {}, acc01 = {}, acc10 = {}, acc11 = {};
  acc00 = wmma_f32_f16(a0, b0, acc00);
  acc01 = wmma_f32_f16(a0, b1, acc01);
  acc10 = wmma_f32_f16(a1, b0, acc10);
  acc11 = wmma_f32_f16(a1, b1, acc11);

  const float scale = 0.0625f;              // 1/sqrt(256)
  const int ib = bI + wm * 32;
  const int jb = bJ + wn * 32;
#pragma unroll
  for (int r = 0; r < 8; ++r) {
    const int rr = r + 8 * hf;
    Sc[(long)(ib + rr)      * Wd + jb + lr]      = acc00[r] * scale;
    Sc[(long)(ib + rr)      * Wd + jb + 16 + lr] = acc01[r] * scale;
    Sc[(long)(ib + 16 + rr) * Wd + jb + lr]      = acc10[r] * scale;
    Sc[(long)(ib + 16 + rr) * Wd + jb + 16 + lr] = acc11[r] * scale;
  }
}

// ---------------------------------------------------------------------------
// Fused softmax + P@V; one wave per (bch, 16-row tile).  Row max/sum scans and
// fragment loads are b128-vectorized; exp(S-m) f16 A-fragments built on the
// fly vs V B-fragments (2 WMMA per 32-wide j-step).
// ---------------------------------------------------------------------------
__global__ void softmax_pv_kernel(const float* __restrict__ S,
                                  const float* __restrict__ V,
                                  float* __restrict__ A) {
  const int wave = threadIdx.x >> 5;
  const int lane = threadIdx.x & 31;
  const int unit  = blockIdx.x * 8 + wave;    // [0, B*C*H*32)
  const int itile = unit & 31;
  const int bch   = unit >> 5;
  const int hd = bch & (Hh - 1);
  const int bc = bch >> 3;
  const float* Sc = S + (long)bch * Wd * Wd;
  const float* Vc = V + ((long)bc * Ff + (long)hd * DH) * Wd;
  float*       Ac = A + ((long)bc * Ff + (long)hd * DH) * Wd;

  const int row = lane & 15;
  const int seg = lane >> 4;
  const float4* S4 = (const float4*)(Sc + (long)(itile * 16 + row) * Wd +
                                     seg * (Wd / 2));

  float m = -3.0e38f;
  for (int j = 0; j < Wd / 8; ++j) {
    const float4 t = S4[j];
    m = fmaxf(m, fmaxf(fmaxf(t.x, t.y), fmaxf(t.z, t.w)));
  }
  m = fmaxf(m, __shfl_xor(m, 16, 32));
  float s = 0.0f;
  for (int j = 0; j < Wd / 8; ++j) {
    const float4 t = S4[j];
    s += __expf(t.x - m) + __expf(t.y - m) + __expf(t.z - m) + __expf(t.w - m);
  }
  s += __shfl_xor(s, 16, 32);

  const int hf = seg;
  v8f acc0 = {}, acc1 = {};
  const float* Sr = Sc + (long)(itile * 16 + row) * Wd;
  for (int jt = 0; jt < Wd / 32; ++jt) {
    // A[i][k] = exp(S[i][jt*32+k]-m); k = hf*16+e contiguous -> 4x b128
    const float4 s0 = *(const float4*)(Sr + jt * 32 + hf * 16);
    const float4 s1 = *(const float4*)(Sr + jt * 32 + hf * 16 + 4);
    const float4 s2 = *(const float4*)(Sr + jt * 32 + hf * 16 + 8);
    const float4 s3 = *(const float4*)(Sr + jt * 32 + hf * 16 + 12);
    v16h a;
    a[0]  = (_Float16)__expf(s0.x - m); a[1]  = (_Float16)__expf(s0.y - m);
    a[2]  = (_Float16)__expf(s0.z - m); a[3]  = (_Float16)__expf(s0.w - m);
    a[4]  = (_Float16)__expf(s1.x - m); a[5]  = (_Float16)__expf(s1.y - m);
    a[6]  = (_Float16)__expf(s1.z - m); a[7]  = (_Float16)__expf(s1.w - m);
    a[8]  = (_Float16)__expf(s2.x - m); a[9]  = (_Float16)__expf(s2.y - m);
    a[10] = (_Float16)__expf(s2.z - m); a[11] = (_Float16)__expf(s2.w - m);
    a[12] = (_Float16)__expf(s3.x - m); a[13] = (_Float16)__expf(s3.y - m);
    a[14] = (_Float16)__expf(s3.z - m); a[15] = (_Float16)__expf(s3.w - m);

    // B[k][n] = v[(g0+n)*Wd + jt*32+k]; k = hf*16+e contiguous -> 4x b128
    const float* vb0 = Vc + (long)row * Wd + jt * 32 + hf * 16;
    const float* vb1 = vb0 + 16L * Wd;
    v16h b0, b1;
#pragma unroll
    for (int q = 0; q < 4; ++q) {
      const float4 t0 = *(const float4*)(vb0 + 4 * q);
      const float4 t1 = *(const float4*)(vb1 + 4 * q);
      b0[4 * q + 0] = (_Float16)t0.x; b0[4 * q + 1] = (_Float16)t0.y;
      b0[4 * q + 2] = (_Float16)t0.z; b0[4 * q + 3] = (_Float16)t0.w;
      b1[4 * q + 0] = (_Float16)t1.x; b1[4 * q + 1] = (_Float16)t1.y;
      b1[4 * q + 2] = (_Float16)t1.z; b1[4 * q + 3] = (_Float16)t1.w;
    }
    acc0 = wmma_f32_f16(a, b0, acc0);
    acc1 = wmma_f32_f16(a, b1, acc1);
  }
#pragma unroll
  for (int r = 0; r < 8; ++r) {
    const int   ir  = r + 8 * hf;
    const float inv = 1.0f / __shfl(s, ir, 32);
    const long  io  = (long)(itile * 16 + ir);
    Ac[(long)row * Wd + io]        = acc0[r] * inv;
    Ac[((long)row + 16) * Wd + io] = acc1[r] * inv;
  }
}

// ---------------------------------------------------------------------------
// Depthwise 1x1 cross-channel mix (+ SquaredReLU or residual-add).
// ---------------------------------------------------------------------------
__global__ void dwmix_kernel(const float* __restrict__ in,
                             const float* __restrict__ wd,
                             const float* __restrict__ resid,
                             float* __restrict__ out,
                             long FW, int sqrelu, long total) {
  const long idx = (long)blockIdx.x * blockDim.x + threadIdx.x;
  if (idx >= total) return;
  const long fw = idx % FW;
  const long t  = idx / FW;
  const int  o  = (int)(t % Cc);
  const long bb = t / Cc;
  const float* base = in + (bb * Cc) * FW + fw;
  float acc = 0.0f;
#pragma unroll
  for (int c2 = 0; c2 < Cc; ++c2) acc += wd[o * Cc + c2] * base[(long)c2 * FW];
  if (sqrelu) {
    const float r = fmaxf(acc, 0.0f);
    out[idx] = r * r;
  } else {
    out[idx] = acc + (resid ? resid[idx] : 0.0f);
  }
}

__global__ void add_kernel(const float* __restrict__ a,
                           const float* __restrict__ b,
                           float* __restrict__ out, long n) {
  const long idx = (long)blockIdx.x * blockDim.x + threadIdx.x;
  if (idx < n) out[idx] = a[idx] + b[idx];
}

// ---------------------------------------------------------------------------
extern "C" void kernel_launch(void* const* d_in, const int* in_sizes, int n_in,
                              void* d_out, int out_size, void* d_ws,
                              size_t ws_size, hipStream_t stream) {
  (void)in_sizes; (void)n_in; (void)out_size; (void)ws_size;
  const float* x      = (const float*)d_in[0];
  const float* conv_q = (const float*)d_in[1];
  const float* conv_k = (const float*)d_in[2];
  const float* conv_v = (const float*)d_in[3];
  const float* wq     = (const float*)d_in[4];
  const float* wk     = (const float*)d_in[5];
  const float* wv     = (const float*)d_in[6];
  const float* wo_pw  = (const float*)d_in[7];
  const float* wo_dw  = (const float*)d_in[8];
  const float* g1     = (const float*)d_in[9];
  const float* b1     = (const float*)d_in[10];
  const float* g2     = (const float*)d_in[11];
  const float* b2     = (const float*)d_in[12];
  const float* w1_pw  = (const float*)d_in[13];
  const float* w1_dw  = (const float*)d_in[14];
  const float* w2_pw  = (const float*)d_in[15];

  float* out_h  = (float*)d_out;        // (b,c,f,w)   = S1 floats
  float* out_qk = out_h + S1;           // (b,c,H,w,w) = 16*S1 floats

  float* ws   = (float*)d_ws;
  float* n1   = ws + 0 * S1;            // LN output (reused for LN2)
  float* cbuf = ws + 1 * S1;            // conv output (reused q/k/v)
  float* qb   = ws + 2 * S1;
  float* kb   = ws + 3 * S1;
  float* vb   = ws + 4 * S1;
  float* atb  = ws + 5 * S1;            // attention output, (b,c,f,w)
  float* yb   = ws + 6 * S1;            // pw GEMM out (reused for FFN z)
  float* hb   = ws + 7 * S1;            // x + attn residual
  float* ub   = ws + 8 * S1;            // FFN hidden, 4*S1
  float* u2b  = ws + 12 * S1;           // mixed + SquaredReLU, 4*S1

  const dim3 blk(256);

  // --- attention ---
  ln_kernel<<<dim3((Bb * Cc * Wd) / 8), blk, 0, stream>>>(x, g1, b1, n1);

  conv3x3_kernel<<<dim3((unsigned)(S1 / 256)), blk, 0, stream>>>(n1, conv_q, cbuf);
  gemm_pc_kernel<Ff, Ff, Wd><<<dim3(16, Bb * Cc), blk, 0, stream>>>(wq, cbuf, qb);
  conv3x3_kernel<<<dim3((unsigned)(S1 / 256)), blk, 0, stream>>>(n1, conv_k, cbuf);
  gemm_pc_kernel<Ff, Ff, Wd><<<dim3(16, Bb * Cc), blk, 0, stream>>>(wk, cbuf, kb);
  conv3x3_kernel<<<dim3((unsigned)(S1 / 256)), blk, 0, stream>>>(n1, conv_v, cbuf);
  gemm_pc_kernel<Ff, Ff, Wd><<<dim3(16, Bb * Cc), blk, 0, stream>>>(wv, cbuf, vb);

  scores_kernel<<<dim3(32, Bb * Cc * Hh), blk, 0, stream>>>(qb, kb, out_qk);
  softmax_pv_kernel<<<dim3((Bb * Cc * Hh * 32) / 8), blk, 0, stream>>>(out_qk, vb, atb);

  gemm_pc_kernel<Ff, Ff, Wd><<<dim3(16, Bb * Cc), blk, 0, stream>>>(wo_pw, atb, yb);
  dwmix_kernel<<<dim3((unsigned)(S1 / 256)), blk, 0, stream>>>(
      yb, wo_dw, x, hb, (long)Ff * Wd, 0, S1);

  // --- FFN ---
  ln_kernel<<<dim3((Bb * Cc * Wd) / 8), blk, 0, stream>>>(hb, g2, b2, n1);
  gemm_pc_kernel<Ee, Ff, Wd><<<dim3(64, Bb * Cc), blk, 0, stream>>>(w1_pw, n1, ub);
  dwmix_kernel<<<dim3((unsigned)((4 * S1) / 256)), blk, 0, stream>>>(
      ub, w1_dw, nullptr, u2b, (long)Ee * Wd, 1, 4 * S1);
  gemm_pc_kernel<Ff, Ee, Wd><<<dim3(16, Bb * Cc), blk, 0, stream>>>(w2_pw, u2b, yb);
  add_kernel<<<dim3((unsigned)(S1 / 256)), blk, 0, stream>>>(hb, yb, out_h, S1);
}
